// TriangleMultiplicationXSMM_52948356825352
// MI455X (gfx1250) — compile-verified, matrix operand order checked by reading, not run on hardware
//
#include <hip/hip_runtime.h>
#include <hip/hip_bf16.h>

#define NN 512
#define CZ 128
#define C2 256
#define LN_EPS 1e-5f

typedef __attribute__((ext_vector_type(16))) __bf16 v16bf;
typedef __attribute__((ext_vector_type(8)))  float  v8f;

// ---------------------------------------------------------------------------
// Helpers
// ---------------------------------------------------------------------------

// Async DMA of 16 bytes global -> LDS (GLOBAL_LOAD_ASYNC_TO_LDS_B128,
// tracked by ASYNCcnt).  LDS operand is the 32-bit LDS byte address
// (low 32 bits of the generic shared-aperture pointer, ISA 10.2).
__device__ inline void async_copy_b128(const void* gptr, void* lptr) {
  uint32_t l = (uint32_t)(uintptr_t)lptr;
  uint64_t g = (uint64_t)(uintptr_t)gptr;
  asm volatile("global_load_async_to_lds_b128 %0, %1, off"
               :: "v"(l), "v"(g) : "memory");
}

__device__ inline void wait_async_le4() {
#if __has_builtin(__builtin_amdgcn_s_wait_asynccnt)
  __builtin_amdgcn_s_wait_asynccnt(4);
#else
  asm volatile("s_wait_asynccnt 0x4" ::: "memory");
#endif
}
__device__ inline void wait_async_0() {
#if __has_builtin(__builtin_amdgcn_s_wait_asynccnt)
  __builtin_amdgcn_s_wait_asynccnt(0);
#else
  asm volatile("s_wait_asynccnt 0x0" ::: "memory");
#endif
}

// Load a 16x16 B-fragment (K=32 slice) from a row-major fp32 weight W[d][k]
// (ld = 128), converting to bf16.  B lane layout mirrored from ISA 7.12.2:
// lane n = lane&15, k = 32*kstep + (h&7) + (h>=8?16:0) + (lane>=16?8:0).
__device__ inline v16bf load_w_frag(const float* __restrict__ W, int d0,
                                    int kstep, int lane) {
  const float* row = W + (size_t)(d0 + (lane & 15)) * CZ;
  const int kb = kstep * 32 + ((lane & 16) ? 8 : 0);
  v16bf b;
#pragma unroll
  for (int h = 0; h < 8; ++h) {
    b[h]     = (__bf16)row[kb + h];
    b[h + 8] = (__bf16)row[kb + 16 + h];
  }
  return b;
}

// Load a 16x32 A-fragment from an LDS row (row-major bf16, lane row = lane&15).
__device__ inline v16bf load_lds_frag(const __bf16* __restrict__ row, int lane) {
  const int kb = (lane & 16) ? 8 : 0;
  v16bf a;
#pragma unroll
  for (int h = 0; h < 8; ++h) {
    a[h]     = row[kb + h];
    a[h + 8] = row[kb + 16 + h];
  }
  return a;
}

__device__ inline float sigmoidf_(float x) {
  return 1.0f / (1.0f + __expf(-x));
}

// LayerNorm a [16][128] fp32 tile (contiguous in global at src) into bf16 LDS.
// 256 threads; leaves dst ready after final barrier.
__device__ inline void ln_tile_to_bf16(const float* __restrict__ src,
                                       const float* __restrict__ w,
                                       const float* __restrict__ b,
                                       float (*xs)[CZ],
                                       float (*red)[16][2],
                                       float (*stats)[2],
                                       __bf16 (*dst)[CZ], int t) {
  for (int idx = t; idx < 16 * CZ / 4; idx += 256)
    ((float4*)&xs[0][0])[idx] = ((const float4*)src)[idx];
  __syncthreads();

  const int r = t >> 4, seg = t & 15;
  {
    float s = 0.f, s2 = 0.f;
#pragma unroll
    for (int u = 0; u < 8; ++u) {
      float v = xs[r][seg * 8 + u];
      s += v; s2 += v * v;
    }
    red[r][seg][0] = s; red[r][seg][1] = s2;
  }
  __syncthreads();
  if (t < 16) {
    float s = 0.f, s2 = 0.f;
#pragma unroll
    for (int u = 0; u < 16; ++u) { s += red[t][u][0]; s2 += red[t][u][1]; }
    float mu  = s  * (1.0f / CZ);
    float var = s2 * (1.0f / CZ) - mu * mu;
    stats[t][0] = mu;
    stats[t][1] = rsqrtf(var + LN_EPS);
  }
  __syncthreads();
  {
    const float mu = stats[r][0], rs = stats[r][1];
#pragma unroll
    for (int u = 0; u < 8; ++u) {
      int c = seg * 8 + u;
      dst[r][c] = (__bf16)((xs[r][c] - mu) * rs * w[c] + b[c]);
    }
  }
  __syncthreads();
}

// ---------------------------------------------------------------------------
// Kernel 1: left-LN + projection/gate GEMMs + mask * sigmoid gating.
// Writes left[c][i][k] and right[c][j][k] (bf16, row-contiguous in k).
// ---------------------------------------------------------------------------
__global__ __launch_bounds__(256) void k_proj(
    const float* __restrict__ act, const float* __restrict__ mask,
    const float* __restrict__ lnw, const float* __restrict__ lnb,
    const float* __restrict__ projW, const float* __restrict__ gateW,
    __bf16* __restrict__ leftW, __bf16* __restrict__ rightW) {
  __shared__ float  xs[16][CZ];
  __shared__ float  red[16][16][2];
  __shared__ float  stats[16][2];
  __shared__ __bf16 xbf[16][CZ];
  __shared__ float  maskrow[16];

  const int t  = threadIdx.x;
  const int p0 = blockIdx.x * 16;

  if (t < 16) maskrow[t] = mask[p0 + t];
  ln_tile_to_bf16(act + (size_t)p0 * CZ, lnw, lnb, xs, red, stats, xbf, t);

  const int wave = t >> 5, lane = t & 31;
  const int m = lane & 15;

  v16bf afrag[4];
#pragma unroll
  for (int s = 0; s < 4; ++s)
    afrag[s] = load_lds_frag(&xbf[m][s * 32], lane);

  const v8f zero = {};
#pragma unroll
  for (int q = 0; q < 2; ++q) {
    const int d0 = (wave * 2 + q) * 16;
    v8f accP = zero, accG = zero;
#pragma unroll
    for (int s = 0; s < 4; ++s) {
      v16bf bp = load_w_frag(projW, d0, s, lane);
      accP = __builtin_amdgcn_wmma_f32_16x16x32_bf16(false, afrag[s], false, bp,
                                                     (short)0, accP, false, false);
      v16bf bg = load_w_frag(gateW, d0, s, lane);
      accG = __builtin_amdgcn_wmma_f32_16x16x32_bf16(false, afrag[s], false, bg,
                                                     (short)0, accG, false, false);
    }
    const int n  = lane & 15;
    const int d  = d0 + n;
    const int hi = lane >> 4;
#pragma unroll
    for (int r = 0; r < 8; ++r) {
      const int mm = r + 8 * hi;
      const int p  = p0 + mm;
      const float val = accP[r] * maskrow[mm] * sigmoidf_(accG[r]);
      const int pi = p >> 9, pj = p & (NN - 1);
      if (d < CZ)
        leftW[(size_t)d * NN * NN + (size_t)pi * NN + pj] = (__bf16)val;
      else
        rightW[(size_t)(d - CZ) * NN * NN + (size_t)pi * NN + pj] = (__bf16)val;
    }
  }
}

// ---------------------------------------------------------------------------
// Kernel 2: per-channel triangle GEMM  y_c = L_c (512x512) * R_c^T (512x512).
// Both operands stored [row][k] bf16 so LDS staging needs no transpose.
// 256 threads compute a 128x128 tile; waves in 2x4 layout each own a
// 64x32 subtile = 4x2 WMMA tiles; K loop of 32, double-buffered LDS fed
// by GLOBAL_LOAD_ASYNC_TO_LDS_B128 (ASYNCcnt) so DMA of slab n+1 overlaps
// WMMA on slab n.
// ---------------------------------------------------------------------------
__global__ __launch_bounds__(256) void k_tri(
    const __bf16* __restrict__ leftW, const __bf16* __restrict__ rightW,
    float* __restrict__ y) {
  const int c  = blockIdx.z;
  const int i0 = blockIdx.x * 128;
  const int j0 = blockIdx.y * 128;
  const __bf16* A = leftW  + (size_t)c * NN * NN;  // [i][k]
  const __bf16* B = rightW + (size_t)c * NN * NN;  // [j][k]

  __shared__ __bf16 As[2][128][40];
  __shared__ __bf16 Bs[2][128][40];

  const int t = threadIdx.x;
  const int wave = t >> 5, lane = t & 31;
  const int wm = wave >> 2;   // 0..1 : 64-row block
  const int wn = wave & 3;    // 0..3 : 32-col block
  const int m  = lane & 15;

  const v8f zero = {};
  v8f acc[4][2];
#pragma unroll
  for (int a = 0; a < 4; ++a)
#pragma unroll
    for (int b = 0; b < 2; ++b) acc[a][b] = zero;

  const int lrow = t >> 1;          // 0..127
  const int loff = (t & 1) * 16;    // bf16 offset within 32-wide slab

  const __bf16* gArow = A + (size_t)(i0 + lrow) * NN + loff;
  const __bf16* gBrow = B + (size_t)(j0 + lrow) * NN + loff;

  // Prologue: async-stage K-slab 0 into buffer 0 (4 async b128 per thread,
  // each copies 16B and its pair at +16B covers the 32B this thread owns).
  async_copy_b128(gArow,     &As[0][lrow][loff]);
  async_copy_b128(gArow + 8, &As[0][lrow][loff + 8]);
  async_copy_b128(gBrow,     &Bs[0][lrow][loff]);
  async_copy_b128(gBrow + 8, &Bs[0][lrow][loff + 8]);

  int cur = 0;
  for (int k0 = 0; k0 < NN; k0 += 32, cur ^= 1) {
    const bool more = (k0 + 32) < NN;
    if (more) {
      const int nxt = cur ^ 1;
      const __bf16* gA = gArow + (k0 + 32);
      const __bf16* gB = gBrow + (k0 + 32);
      async_copy_b128(gA,     &As[nxt][lrow][loff]);
      async_copy_b128(gA + 8, &As[nxt][lrow][loff + 8]);
      async_copy_b128(gB,     &Bs[nxt][lrow][loff]);
      async_copy_b128(gB + 8, &Bs[nxt][lrow][loff + 8]);
      wait_async_le4();   // current slab's 4 DMAs done; prefetch stays in flight
    } else {
      wait_async_0();
    }
    __syncthreads();

    v16bf af[4], bfg[2];
#pragma unroll
    for (int a = 0; a < 4; ++a)
      af[a] = load_lds_frag(&As[cur][wm * 64 + a * 16 + m][0], lane);
#pragma unroll
    for (int b = 0; b < 2; ++b)
      bfg[b] = load_lds_frag(&Bs[cur][wn * 32 + b * 16 + m][0], lane);

#pragma unroll
    for (int a = 0; a < 4; ++a)
#pragma unroll
      for (int b = 0; b < 2; ++b)
        acc[a][b] = __builtin_amdgcn_wmma_f32_16x16x32_bf16(
            false, af[a], false, bfg[b], (short)0, acc[a][b], false, false);

    __syncthreads();  // all waves done with 'cur' before it is re-filled
  }

  const int n = lane & 15, hi = lane >> 4;
#pragma unroll
  for (int a = 0; a < 4; ++a)
#pragma unroll
    for (int b = 0; b < 2; ++b)
#pragma unroll
      for (int r = 0; r < 8; ++r) {
        const int i = i0 + wm * 64 + a * 16 + r + 8 * hi;
        const int j = j0 + wn * 32 + b * 16 + n;
        y[((size_t)i * NN + j) * CZ + c] = acc[a][b][r];
      }
}

// ---------------------------------------------------------------------------
// Kernel 3: center-LN(y) -> output projection; left-LN(act) -> gating linear;
// out = proj * sigmoid(gating).
// ---------------------------------------------------------------------------
__global__ __launch_bounds__(256) void k_out(
    const float* __restrict__ y, const float* __restrict__ act,
    const float* __restrict__ cnw, const float* __restrict__ cnb,
    const float* __restrict__ lnw, const float* __restrict__ lnb,
    const float* __restrict__ outW, const float* __restrict__ gatW,
    float* __restrict__ out) {
  __shared__ float  xs[16][CZ];
  __shared__ float  red[16][16][2];
  __shared__ float  stats[16][2];
  __shared__ __bf16 a1[16][CZ];  // center-LN(y)
  __shared__ __bf16 a2[16][CZ];  // left-LN(act)

  const int t  = threadIdx.x;
  const int p0 = blockIdx.x * 16;

  ln_tile_to_bf16(y   + (size_t)p0 * CZ, cnw, cnb, xs, red, stats, a1, t);
  ln_tile_to_bf16(act + (size_t)p0 * CZ, lnw, lnb, xs, red, stats, a2, t);

  const int wave = t >> 5, lane = t & 31;
  const int m = lane & 15;
  const int d0 = wave * 16;

  const v8f zero = {};
  v8f accO = zero, accG = zero;
#pragma unroll
  for (int s = 0; s < 4; ++s) {
    v16bf f1 = load_lds_frag(&a1[m][s * 32], lane);
    v16bf bo = load_w_frag(outW, d0, s, lane);
    accO = __builtin_amdgcn_wmma_f32_16x16x32_bf16(false, f1, false, bo,
                                                   (short)0, accO, false, false);
    v16bf f2 = load_lds_frag(&a2[m][s * 32], lane);
    v16bf bg = load_w_frag(gatW, d0, s, lane);
    accG = __builtin_amdgcn_wmma_f32_16x16x32_bf16(false, f2, false, bg,
                                                   (short)0, accG, false, false);
  }

  const int n = lane & 15, hi = lane >> 4;
#pragma unroll
  for (int r = 0; r < 8; ++r) {
    const int p = p0 + r + 8 * hi;
    out[(size_t)p * CZ + d0 + n] = accO[r] * sigmoidf_(accG[r]);
  }
}

// ---------------------------------------------------------------------------
// Launcher
// ---------------------------------------------------------------------------
extern "C" void kernel_launch(void* const* d_in, const int* in_sizes, int n_in,
                              void* d_out, int out_size, void* d_ws,
                              size_t ws_size, hipStream_t stream) {
  (void)in_sizes; (void)n_in; (void)out_size; (void)ws_size;

  const float* act   = (const float*)d_in[0];
  const float* mask  = (const float*)d_in[1];
  const float* lnw   = (const float*)d_in[2];
  const float* lnb   = (const float*)d_in[3];
  const float* projW = (const float*)d_in[4];  // [256,128]
  const float* gateW = (const float*)d_in[5];  // [256,128]
  const float* cnw   = (const float*)d_in[6];
  const float* cnb   = (const float*)d_in[7];
  const float* outW  = (const float*)d_in[8];  // [128,128]
  const float* gatW  = (const float*)d_in[9];  // [128,128]
  float* out = (float*)d_out;

  char* ws = (char*)d_ws;
  __bf16* leftW  = (__bf16*)(ws);                           // 64 MiB
  __bf16* rightW = (__bf16*)(ws + ((size_t)64 << 20));      // 64 MiB
  float*  yv     = (float*) (ws + ((size_t)128 << 20));     // 128 MiB

  k_proj<<<NN * NN / 16, 256, 0, stream>>>(act, mask, lnw, lnb, projW, gateW,
                                           leftW, rightW);
  k_tri<<<dim3(NN / 128, NN / 128, CZ), 256, 0, stream>>>(leftW, rightW, yv);
  k_out<<<NN * NN / 16, 256, 0, stream>>>(yv, act, cnw, cnb, lnw, lnb, outW,
                                          gatW, out);
}
